// LSTM_61615600828783
// MI455X (gfx1250) — compile-verified, matrix operand order checked by reading, not run on hardware
//
#include <hip/hip_runtime.h>
#include <hip/hip_bf16.h>

// ---------------------------------------------------------------------------
// Persistent fused ReLU+LSTM for MI455X (gfx1250).
//   gates_t = [relu(x_t), h_t] @ [W_ih ; W_hh]^T + (b_ih + b_hh)
// 8 workgroups x 512 threads (16 wave32). WG w owns h columns [32w,32w+32):
// gate rows {g*256 + 32w .. +31 | g=0..3} = 128 rows of the combined
// [1024 x 512] weight, pre-swizzled into per-wave WMMA B-fragments that
// live in VGPRs for all 2048 timesteps. Per step: stage A = [relu(x_t)|h_t]
// (32x512 bf16) in LDS, each wave runs 16 v_wmma_f32_16x16x32_bf16 split
// over TWO accumulators (halves the XDL dependency chain on the sequential
// critical path), then the cell update runs elementwise; h is exchanged
// between WGs through a double-buffered bf16 buffer in d_ws guarded by a
// per-step release/acquire arrival counter (software grid barrier).
// ---------------------------------------------------------------------------

typedef __attribute__((ext_vector_type(16))) __bf16 v16bf;
typedef __attribute__((ext_vector_type(8)))  float  v8f;

namespace {
constexpr int BB   = 32;    // batch (GEMM M)
constexpr int TT   = 2048;  // timesteps
constexpr int DD   = 256;   // input dim
constexpr int HH   = 256;   // hidden dim
constexpr int KTOT = DD + HH;          // 512 = GEMM K
constexpr int NWG  = 8;                // workgroups
constexpr int NPW  = 4 * HH / NWG;     // 128 gate rows per WG
constexpr int NTH  = 512;              // threads per WG (16 waves)
constexpr int KC   = KTOT / 32;        // 16 wmma k-chunks
constexpr int APAD = 8;                // LDS pad (keeps 16B row alignment)
constexpr int GPAD = 4;
}

#if defined(__has_builtin)
#if __has_builtin(__builtin_amdgcn_cvt_pk_bf16_f32)
#define LSTM_HAVE_PK_BF16 1
#endif
#endif

__device__ __forceinline__ unsigned short f32_to_bf16_bits(float f) {
  union { float f; unsigned u; } v; v.f = f;
  unsigned r = v.u + 0x7FFFu + ((v.u >> 16) & 1u);   // round-to-nearest-even
  return (unsigned short)(r >> 16);
}
__device__ __forceinline__ unsigned pack_bf16x2(float lo, float hi) {
#ifdef LSTM_HAVE_PK_BF16
  auto p = __builtin_amdgcn_cvt_pk_bf16_f32(lo, hi);  // v_cvt_pk_bf16_f32
  unsigned u;
  __builtin_memcpy(&u, &p, sizeof(u));
  return u;
#else
  return (unsigned)f32_to_bf16_bits(lo) | ((unsigned)f32_to_bf16_bits(hi) << 16);
#endif
}
__device__ __forceinline__ float sigm(float x) { return 1.0f / (1.0f + __expf(-x)); }
__device__ __forceinline__ float tanh_fast(float x) { return 1.0f - 2.0f / (1.0f + __expf(2.0f * x)); }

__global__ __launch_bounds__(NTH, 1)
void lstm_persistent(const float* __restrict__ x,      // [B,T,D]
                     const float* __restrict__ W_ih,   // [4H,D]
                     const float* __restrict__ W_hh,   // [4H,H]
                     const float* __restrict__ b_ih,   // [4H]
                     const float* __restrict__ b_hh,   // [4H]
                     float* __restrict__ out,          // [B,T,H]
                     int* __restrict__ arrive,         // [T] zeroed each launch
                     unsigned short* __restrict__ hx)  // [2][B][H] bf16, zeroed
{
  __shared__ unsigned short a_buf[BB][KTOT + APAD];  // bf16 A staging (row = batch)
  __shared__ float gate_buf[BB][NPW + GPAD];         // f32 gate outputs
  __shared__ float c_buf[BB][HH / NWG];              // cell state slice

  const int tid   = threadIdx.x;
  const int wg    = blockIdx.x;
  const int wave  = tid >> 5;
  const int lane  = tid & 31;
  const int mtile = wave & 1;     // output rows 0-15 / 16-31
  const int ntile = wave >> 1;    // 0..7 -> 16 gate rows each
  const int gate  = ntile >> 1;   // 0:i 1:f 2:g 3:o
  const int lhalf = lane >> 4;
  const int l16   = lane & 15;

  const int nloc  = ntile * 16 + l16;        // 0..127 within WG slice
  const int col   = wg * 32 + (nloc & 31);   // h column 0..255
  const int rglob = gate * HH + col;         // gate row 0..1023
  const float bias = b_ih[rglob] + b_hh[rglob];

  // ---- Pre-swizzle this wave's B fragments (ISA 7.12.2 16-bit layout):
  // lane l16 = column nn; frag elem j <-> K = kc*32 + (j>>3)*16 + lhalf*8 + (j&7)
  v16bf bfrag[KC];
  {
    const float* wi = W_ih + (size_t)rglob * DD;
    const float* wh = W_hh + (size_t)rglob * HH;
#pragma unroll
    for (int kc = 0; kc < KC; ++kc) {
      union { v16bf v; unsigned short s[16]; } u;
#pragma unroll
      for (int j = 0; j < 16; ++j) {
        int k = kc * 32 + ((j >> 3) * 16) + lhalf * 8 + (j & 7);
        float w = (k < DD) ? wi[k] : wh[k - DD];
        u.s[j] = f32_to_bf16_bits(w);
      }
      bfrag[kc] = u.v;
    }
  }

  // zero cell state
  for (int i = tid; i < BB * (HH / NWG); i += NTH) ((float*)c_buf)[i] = 0.0f;
  __syncthreads();

  for (int t = 0; t < TT; ++t) {
    // ---- stage relu(x_t) -> bf16 LDS (independent of h; overlaps barrier)
    {
      const int b  = tid >> 4;
      const int d0 = (tid & 15) * 16;
      const float4* xp = (const float4*)(x + ((size_t)b * TT + t) * DD + d0);
      float4 f0 = xp[0], f1 = xp[1], f2 = xp[2], f3 = xp[3];
      if (t + 1 < TT)   // pull next step's activations toward L2/WGP$
        __builtin_prefetch(x + ((size_t)b * TT + (t + 1)) * DD + d0, 0, 3);
      unsigned* dst = (unsigned*)&a_buf[b][d0];
      dst[0] = pack_bf16x2(fmaxf(f0.x, 0.f), fmaxf(f0.y, 0.f));
      dst[1] = pack_bf16x2(fmaxf(f0.z, 0.f), fmaxf(f0.w, 0.f));
      dst[2] = pack_bf16x2(fmaxf(f1.x, 0.f), fmaxf(f1.y, 0.f));
      dst[3] = pack_bf16x2(fmaxf(f1.z, 0.f), fmaxf(f1.w, 0.f));
      dst[4] = pack_bf16x2(fmaxf(f2.x, 0.f), fmaxf(f2.y, 0.f));
      dst[5] = pack_bf16x2(fmaxf(f2.z, 0.f), fmaxf(f2.w, 0.f));
      dst[6] = pack_bf16x2(fmaxf(f3.x, 0.f), fmaxf(f3.y, 0.f));
      dst[7] = pack_bf16x2(fmaxf(f3.z, 0.f), fmaxf(f3.w, 0.f));
    }

    // ---- grid barrier: wait until all 8 WGs published h_t (skip t==0; h_0=0)
    if (t > 0) {
      if (tid == 0) {
        while (__hip_atomic_load(&arrive[t - 1], __ATOMIC_ACQUIRE,
                                 __HIP_MEMORY_SCOPE_AGENT) < NWG)
          __builtin_amdgcn_s_sleep(2);
      }
      __syncthreads();
    }

    // ---- stage h_t (bf16) from exchange buffer hx[t&1]
    {
      const int b  = tid >> 4;
      const int k0 = (tid & 15) * 16;
      const uint4* hp = (const uint4*)(hx + ((size_t)(t & 1) * BB + b) * HH + k0);
      uint4 q0 = hp[0], q1 = hp[1];
      *(uint4*)&a_buf[b][DD + k0]     = q0;
      *(uint4*)&a_buf[b][DD + k0 + 8] = q1;
    }
    __syncthreads();

    // ---- WMMA: acc[16x16] += A[16x32] * B[32x16], 16 k-chunks over
    //      two accumulators (halves D->C dependency chain in the XDL pipe)
    v8f acc0 = {}, acc1 = {};
    {
      const int m = mtile * 16 + l16;   // A row = batch index (lane layout)
#pragma unroll
      for (int kc = 0; kc < KC; kc += 2) {
        union { v16bf v; uint4 q[2]; } a0, a1;
        a0.q[0] = *(const uint4*)&a_buf[m][kc * 32 + lhalf * 8];
        a0.q[1] = *(const uint4*)&a_buf[m][kc * 32 + 16 + lhalf * 8];
        a1.q[0] = *(const uint4*)&a_buf[m][(kc + 1) * 32 + lhalf * 8];
        a1.q[1] = *(const uint4*)&a_buf[m][(kc + 1) * 32 + 16 + lhalf * 8];
        acc0 = __builtin_amdgcn_wmma_f32_16x16x32_bf16(
            false, a0.v, false, bfrag[kc], (short)0, acc0, false, false);
        acc1 = __builtin_amdgcn_wmma_f32_16x16x32_bf16(
            false, a1.v, false, bfrag[kc + 1], (short)0, acc1, false, false);
      }
    }

    // ---- scatter gates (+bias) to LDS (C/D layout: VGPR v -> M = v + lhalf*8)
#pragma unroll
    for (int v = 0; v < 8; ++v)
      gate_buf[mtile * 16 + lhalf * 8 + v][nloc] = (acc0[v] + acc1[v]) + bias;
    __syncthreads();

    // ---- LSTM cell update: 32 batches x 32 owned columns, 2 per thread
#pragma unroll
    for (int r = 0; r < 2; ++r) {
      const int idx = tid + r * NTH;
      const int b = idx >> 5;
      const int j = idx & 31;
      float gi = sigm(gate_buf[b][j]);
      float gf = sigm(gate_buf[b][32 + j]);
      float gg = tanh_fast(gate_buf[b][64 + j]);
      float go = sigm(gate_buf[b][96 + j]);
      float c  = gf * c_buf[b][j] + gi * gg;
      c_buf[b][j] = c;
      float h = go * tanh_fast(c);
      const int jc = wg * 32 + j;
      out[((size_t)b * TT + t) * HH + jc] = h;
      hx[((size_t)((t + 1) & 1) * BB + b) * HH + jc] = f32_to_bf16_bits(h);
    }

    // ---- publish h_{t+1} and signal arrival for step t
    __threadfence();
    __syncthreads();
    if (tid == 0)
      __hip_atomic_fetch_add(&arrive[t], 1, __ATOMIC_RELEASE,
                             __HIP_MEMORY_SCOPE_AGENT);
    __syncthreads();   // a_buf/gate_buf safe to overwrite next iteration
  }
}

extern "C" void kernel_launch(void* const* d_in, const int* in_sizes, int n_in,
                              void* d_out, int out_size, void* d_ws, size_t ws_size,
                              hipStream_t stream) {
  const float* x    = (const float*)d_in[0];
  const float* W_ih = (const float*)d_in[1];
  const float* W_hh = (const float*)d_in[2];
  const float* b_ih = (const float*)d_in[3];
  const float* b_hh = (const float*)d_in[4];
  float* out = (float*)d_out;

  // workspace layout: [0,8K) arrival counters (T ints), [8K, 8K+32K) bf16 h x2
  int* arrive = (int*)d_ws;
  unsigned short* hx = (unsigned short*)((char*)d_ws + 8192);
  size_t clear_bytes = 8192 + (size_t)2 * BB * HH * sizeof(unsigned short);
  hipMemsetAsync(d_ws, 0, clear_bytes, stream);   // graph-capture safe

  lstm_persistent<<<dim3(NWG), dim3(NTH), 0, stream>>>(
      x, W_ih, W_hh, b_ih, b_hh, out, arrive, hx);
}
